// SubgraphEmbeddingRegressorModel_10557029613659
// MI455X (gfx1250) — matrix operand order repaired and checked by reading, not compiled, not run on hardware
//
#include <hip/hip_runtime.h>

#define NN 50000
#define NE 800000
#define NG 2000
#define NP 4096
#define INC 64
#define EMBF 128
#define HIDF 256

typedef __attribute__((ext_vector_type(16))) _Float16 v16h;
typedef __attribute__((ext_vector_type(8)))  float    v8f;

// ---------------- elementwise / init helpers ----------------
__global__ void fill_f32(float* __restrict__ p, float v, int n) {
  int i = blockIdx.x * blockDim.x + threadIdx.x;
  if (i < n) p[i] = v;
}

__global__ void deg_edges_kernel(const int* __restrict__ col, float* __restrict__ deg, int E) {
  int e = blockIdx.x * blockDim.x + threadIdx.x;
  if (e < E) atomicAdd(&deg[col[e]], 1.0f);
}

__global__ void rsqrt_kernel(const float* __restrict__ deg, float* __restrict__ dinv, int n) {
  int i = blockIdx.x * blockDim.x + threadIdx.x;
  if (i < n) dinv[i] = rsqrtf(deg[i]);
}

// out[n,f] = h[n,f] / deg[n] + b[f]    (self-loop term + bias), F == 128
__global__ void self_bias_kernel(const float* __restrict__ h, const float* __restrict__ deg,
                                 const float* __restrict__ b, float* __restrict__ out, int n) {
  int i = blockIdx.x * blockDim.x + threadIdx.x;
  if (i < n) out[i] = h[i] / deg[i >> 7] + b[i & 127];
}

__global__ void relu_kernel(float* __restrict__ p, int n) {
  int i = blockIdx.x * blockDim.x + threadIdx.x;
  if (i < n) p[i] = fmaxf(p[i], 0.0f);
}

// ---------------- edge aggregation: out[col] += dinv[row]*dinv[col]*h[row] ----------------
// one wave per edge; lanes stride the 128 features (coalesced loads + f32 atomics, L2-resident)
__global__ void edge_agg_kernel(const int* __restrict__ rows, const int* __restrict__ cols,
                                const float* __restrict__ dinv, const float* __restrict__ h,
                                float* __restrict__ out, int E) {
  int wid  = (blockIdx.x * blockDim.x + threadIdx.x) >> 5;
  int lane = threadIdx.x & 31;
  if (wid >= E) return;
  int r = rows[wid], c = cols[wid];
  float w = dinv[r] * dinv[c];
  const float* hp = h + (size_t)r * EMBF;
  float* op = out + (size_t)c * EMBF;
#pragma unroll
  for (int k = 0; k < EMBF / 32; ++k) {
    int f = lane + 32 * k;
    atomicAdd(&op[f], w * hp[f]);
  }
}

// ---------------- WMMA GEMM: D[M,O] = A[M,K] @ W[O,K]^T (+bias)(+relu) ----------------
// one wave32 per 16x16 output tile; K stepped by 32 through v_wmma_f32_16x16x32_f16
__global__ void gemm_wmma_kernel(const float* __restrict__ A, const float* __restrict__ W,
                                 float* __restrict__ D, const float* __restrict__ bias,
                                 int M, int K, int O, int relu) {
  int wave = (blockIdx.x * blockDim.x + threadIdx.x) >> 5;
  int lane = threadIdx.x & 31;
  int otiles = O >> 4;
  int mtile = wave / otiles;
  int otile = wave - mtile * otiles;
  if (mtile * 16 >= M) return;          // wave-uniform: EXEC stays all-ones for WMMA

  int half = lane >> 4;                 // 0: lanes 0-15, 1: lanes 16-31
  int l16  = lane & 15;
  int arow = mtile * 16 + l16;          // A row held by this lane
  int bcol = otile * 16 + l16;          // B column (= row of W) held by this lane
  int ka = half * 8;                    // A K-offset per ISA layout
  int kb = half * 16;                   // unused: B loads use ka, same K order as A

  v8f acc = {};
  for (int k0 = 0; k0 < K; k0 += 32) {
    const float* ap = A + (size_t)arow * K + k0;
    const float* wp = W + (size_t)bcol * K + k0;
    v16h a, b;
#pragma unroll
    for (int e = 0; e < 8; ++e) {
      a[e]     = (_Float16)ap[ka + e];          // K = k0 + ka + e
      a[e + 8] = (_Float16)ap[16 + ka + e];     // K = k0 + 16 + ka + e
    }
#pragma unroll
    for (int e = 0; e < 16; ++e)
    b[e] = (_Float16)wp[(e & 8) * 2 + ka + (e & 7)];  // K = k0 + 2*(e&8) + ka + (e&7)
    acc = __builtin_amdgcn_wmma_f32_16x16x32_f16(false, a, false, b,
                                                 (short)0, acc, false, false);
  }

  int ocol = otile * 16 + l16;
  float bv = bias ? bias[ocol] : 0.0f;
#pragma unroll
  for (int r = 0; r < 8; ++r) {
    int mrow = mtile * 16 + half * 8 + r;       // C/D: VGPR r -> M = r (+8 for hi lanes)
    float v = acc[r] + bv;
    if (relu) v = fmaxf(v, 0.0f);
    D[(size_t)mrow * O + ocol] = v;
  }
}

// ---------------- graph readout (scatter-mean) ----------------
__global__ void readout_kernel(const float* __restrict__ a, const int* __restrict__ batch,
                               float* __restrict__ gsum, float* __restrict__ cnt, int N) {
  int wid  = (blockIdx.x * blockDim.x + threadIdx.x) >> 5;
  int lane = threadIdx.x & 31;
  if (wid >= N) return;
  int g = batch[wid];
  const float* ap = a + (size_t)wid * EMBF;
  float* gp = gsum + (size_t)g * EMBF;
#pragma unroll
  for (int k = 0; k < EMBF / 32; ++k) {
    int f = lane + 32 * k;
    atomicAdd(&gp[f], ap[f]);
  }
  if (lane == 0) atomicAdd(&cnt[g], 1.0f);
}

__global__ void mean_kernel(const float* __restrict__ gsum, const float* __restrict__ cnt,
                            float* __restrict__ ge, int n) {
  int i = blockIdx.x * blockDim.x + threadIdx.x;
  if (i < n) ge[i] = gsum[i] / fmaxf(cnt[i >> 7], 1.0f);
}

// ---------------- np.digitize(dd, unique(dd)+1) == rank among unique ids ----------------
__global__ void rank_kernel(const int* __restrict__ dd, int* __restrict__ rank) {
  __shared__ int pres[NG];
  for (int i = threadIdx.x; i < NG; i += blockDim.x) pres[i] = 0;
  __syncthreads();
  for (int i = threadIdx.x; i < 2 * NP; i += blockDim.x) pres[dd[i]] = 1;
  __syncthreads();
  if (threadIdx.x == 0) {
    int s = 0;
    for (int i = 0; i < NG; ++i) { rank[i] = s; s += pres[i]; }
  }
}

// pairs[p] = concat(ge[rank[dd[0,p]]], ge[rank[dd[1,p]]])
__global__ void pair_gather_kernel(const int* __restrict__ dd, const int* __restrict__ rank,
                                   const float* __restrict__ ge, float* __restrict__ pairs, int P) {
  int wid  = (blockIdx.x * blockDim.x + threadIdx.x) >> 5;
  int lane = threadIdx.x & 31;
  if (wid >= P) return;
  int q0 = rank[dd[wid]];
  int q1 = rank[dd[P + wid]];
  const float* g0 = ge + (size_t)q0 * EMBF;
  const float* g1 = ge + (size_t)q1 * EMBF;
  float* pp = pairs + (size_t)wid * (2 * EMBF);
#pragma unroll
  for (int k = 0; k < EMBF / 32; ++k) {
    int f = lane + 32 * k;
    pp[f]        = g0[f];
    pp[EMBF + f] = g1[f];
  }
}

// out[p] = dot(hid[p,:], w[:]) + b   (final OUT=1 layer, wave reduction)
__global__ void reg2_kernel(const float* __restrict__ hid, const float* __restrict__ w,
                            const float* __restrict__ b, float* __restrict__ out, int P) {
  int wid  = (blockIdx.x * blockDim.x + threadIdx.x) >> 5;
  int lane = threadIdx.x & 31;
  if (wid >= P) return;
  const float* hp = hid + (size_t)wid * HIDF;
  float s = 0.0f;
#pragma unroll
  for (int k = 0; k < HIDF / 32; ++k) {
    int f = lane + 32 * k;
    s += hp[f] * w[f];
  }
#pragma unroll
  for (int off = 16; off > 0; off >>= 1) s += __shfl_down(s, off, 32);
  if (lane == 0) out[wid] = s + b[0];
}

// ---------------- launch ----------------
static inline int ceil_div(long long a, long long b) { return (int)((a + b - 1) / b); }

extern "C" void kernel_launch(void* const* d_in, const int* in_sizes, int n_in,
                              void* d_out, int out_size, void* d_ws, size_t ws_size,
                              hipStream_t stream) {
  const float* x        = (const float*)d_in[0];
  const int*   ei       = (const int*)d_in[1];      // [2, NE]: rows then cols
  const int*   batch    = (const int*)d_in[2];
  const int*   dd       = (const int*)d_in[3];      // [2, NP]
  const float* conv1_w  = (const float*)d_in[4];
  const float* conv1_b  = (const float*)d_in[5];
  const float* conv2_w  = (const float*)d_in[6];
  const float* conv2_b  = (const float*)d_in[7];
  const float* reg1_w   = (const float*)d_in[8];
  const float* reg1_b   = (const float*)d_in[9];
  const float* reg2_w   = (const float*)d_in[10];
  const float* reg2_b   = (const float*)d_in[11];
  float* out = (float*)d_out;
  (void)in_sizes; (void)n_in; (void)out_size; (void)ws_size;

  // workspace layout (floats)
  float* ws = (float*)d_ws;
  size_t o = 0;
  float* bufA  = ws + o; o += (size_t)NN * EMBF;    // h (gemm output)
  float* bufB  = ws + o; o += (size_t)NN * EMBF;    // aggregated / activated
  float* deg   = ws + o; o += NN;
  float* dinv  = ws + o; o += NN;
  float* gsum  = ws + o; o += (size_t)NG * EMBF;
  float* cnt   = ws + o; o += NG;
  float* ge    = ws + o; o += (size_t)NG * EMBF;
  int*   rank  = (int*)(ws + o); o += NG;
  float* pairs = ws + o; o += (size_t)NP * 2 * EMBF;
  float* hid   = ws + o; o += (size_t)NP * HIDF;

  const int B = 256;
  const int nodeF = NN * EMBF;
  const int rows_off = 0, cols_off = NE;

  // degree + normalization
  fill_f32<<<ceil_div(NN, B), B, 0, stream>>>(deg, 1.0f, NN);
  deg_edges_kernel<<<ceil_div(NE, B), B, 0, stream>>>(ei + cols_off, deg, NE);
  rsqrt_kernel<<<ceil_div(NN, B), B, 0, stream>>>(deg, dinv, NN);

  // layer 1: h = x @ W1^T ; out = h/deg + b1 + scatter(norm*h) ; relu
  {
    long long waves = (long long)(NN / 16) * (EMBF / 16);
    gemm_wmma_kernel<<<ceil_div(waves * 32, B), B, 0, stream>>>(
        x, conv1_w, bufA, nullptr, NN, INC, EMBF, 0);
  }
  self_bias_kernel<<<ceil_div(nodeF, B), B, 0, stream>>>(bufA, deg, conv1_b, bufB, nodeF);
  edge_agg_kernel<<<ceil_div((long long)NE * 32, B), B, 0, stream>>>(
      ei + rows_off, ei + cols_off, dinv, bufA, bufB, NE);
  relu_kernel<<<ceil_div(nodeF, B), B, 0, stream>>>(bufB, nodeF);

  // layer 2: h2 = a1 @ W2^T into bufA ; aggregate back into bufB ; relu
  {
    long long waves = (long long)(NN / 16) * (EMBF / 16);
    gemm_wmma_kernel<<<ceil_div(waves * 32, B), B, 0, stream>>>(
        bufB, conv2_w, bufA, nullptr, NN, EMBF, EMBF, 0);
  }
  self_bias_kernel<<<ceil_div(nodeF, B), B, 0, stream>>>(bufA, deg, conv2_b, bufB, nodeF);
  edge_agg_kernel<<<ceil_div((long long)NE * 32, B), B, 0, stream>>>(
      ei + rows_off, ei + cols_off, dinv, bufA, bufB, NE);
  relu_kernel<<<ceil_div(nodeF, B), B, 0, stream>>>(bufB, nodeF);

  // scatter-mean readout
  fill_f32<<<ceil_div(NG * EMBF, B), B, 0, stream>>>(gsum, 0.0f, NG * EMBF);
  fill_f32<<<ceil_div(NG, B), B, 0, stream>>>(cnt, 0.0f, NG);
  readout_kernel<<<ceil_div((long long)NN * 32, B), B, 0, stream>>>(bufB, batch, gsum, cnt, NN);
  mean_kernel<<<ceil_div(NG * EMBF, B), B, 0, stream>>>(gsum, cnt, ge, NG * EMBF);

  // dense id quantization + pair embedding gather
  rank_kernel<<<1, 1024, 0, stream>>>(dd, rank);
  pair_gather_kernel<<<ceil_div((long long)NP * 32, B), B, 0, stream>>>(dd, rank, ge, pairs, NP);

  // regressor: hid = relu(pairs @ reg1_w^T + b1) ; out = hid @ reg2_w^T + b2
  {
    long long waves = (long long)(NP / 16) * (HIDF / 16);
    gemm_wmma_kernel<<<ceil_div(waves * 32, B), B, 0, stream>>>(
        pairs, reg1_w, hid, reg1_b, NP, 2 * EMBF, HIDF, 1);
  }
  reg2_kernel<<<ceil_div((long long)NP * 32, B), B, 0, stream>>>(hid, reg2_w, reg2_b, out, NP);
}
